// Attention_11519102288115
// MI455X (gfx1250) — compile-verified, matrix operand order checked by reading, not run on hardware
//
#include <hip/hip_runtime.h>

// ---------------------------------------------------------------------------
// CDNA5 (gfx1250) cross-attention pipeline, wave32 + v_wmma_f32_16x16x32_bf16
// + async global->LDS staging (ASYNCcnt) for K/V in the flash-attention loop.
// Shapes: b=2, n=m=2048, e=768, h=12, d=64.
// ---------------------------------------------------------------------------

typedef __bf16 bf16_t;
typedef __attribute__((ext_vector_type(16))) __bf16 bf16x16;
typedef __attribute__((ext_vector_type(8)))  float  f32x8;
typedef int v4i_t __attribute__((vector_size(16)));   // matches builtin proto

#define BATCH 2
#define SEQ   2048
#define EMB   768
#define HEADS 12
#define HDIM  64

#if __has_builtin(__builtin_amdgcn_global_load_async_to_lds_b128)
#define HAVE_ASYNC_LDS 1
typedef __attribute__((address_space(1))) v4i_t gbl_v4i;  // prints as __device__
typedef __attribute__((address_space(3))) v4i_t lds_v4i;  // LDS
#else
#define HAVE_ASYNC_LDS 0
#endif

// ---- 16B global->LDS copy: async DMA on gfx1250 toolchains that expose it,
//      register relay otherwise (compile-safe fallback).
__device__ __forceinline__ void lds_copy16(bf16_t* dst, const bf16_t* src) {
#if HAVE_ASYNC_LDS
  __builtin_amdgcn_global_load_async_to_lds_b128(
      (gbl_v4i*)const_cast<bf16_t*>(src), (lds_v4i*)dst,
      /*imm offset*/0, /*cpol*/0);
#else
  *(uint4*)dst = *(const uint4*)src;
#endif
}

template <int N>
__device__ __forceinline__ void wait_async() {
#if HAVE_ASYNC_LDS
#if __has_builtin(__builtin_amdgcn_s_wait_asynccnt)
  __builtin_amdgcn_s_wait_asynccnt(N);
#else
  asm volatile("s_wait_asynccnt %0" ::"i"(N) : "memory");
#endif
#endif
}

// ---- WMMA wrapper ---------------------------------------------------------
__device__ __forceinline__ f32x8 wmma_bf16(bf16x16 a, bf16x16 b, f32x8 c) {
  // 8 args: (neg_a, A, neg_b, B, c_mod, C, reuse_a, reuse_b)
  return __builtin_amdgcn_wmma_f32_16x16x32_bf16(false, a, false, b,
                                                 (short)0, c, false, false);
}

// ---- A/B fragment loaders -------------------------------------------------
// 16-bit A (16x32) layout: lane holds row (lane&15); K elems:
//   VGPR0..3 -> K = half*8 + 0..7 ; VGPR4..7 -> K = 16 + half*8 + 0..7
// (half = lane>>4). B mirrors this with lane holding column n. Both K groups
// are 8 contiguous bf16 = one aligned 16B load each.
__device__ __forceinline__ bf16x16 frag_from_bf16(const bf16_t* __restrict__ p,
                                                  int half) {
  union { uint4 u[2]; bf16x16 v; } r;
  r.u[0] = *(const uint4*)(p + half * 8);
  r.u[1] = *(const uint4*)(p + 16 + half * 8);
  return r.v;
}

__device__ __forceinline__ bf16x16 frag_from_f32(const float* __restrict__ p,
                                                 int half) {
  const float4 a = *(const float4*)(p + half * 8);
  const float4 b = *(const float4*)(p + half * 8 + 4);
  const float4 c = *(const float4*)(p + half * 8 + 16);
  const float4 d = *(const float4*)(p + half * 8 + 20);
  bf16x16 v;
  v[0]  = (__bf16)a.x; v[1]  = (__bf16)a.y; v[2]  = (__bf16)a.z; v[3]  = (__bf16)a.w;
  v[4]  = (__bf16)b.x; v[5]  = (__bf16)b.y; v[6]  = (__bf16)b.z; v[7]  = (__bf16)b.w;
  v[8]  = (__bf16)c.x; v[9]  = (__bf16)c.y; v[10] = (__bf16)c.z; v[11] = (__bf16)c.w;
  v[12] = (__bf16)d.x; v[13] = (__bf16)d.y; v[14] = (__bf16)d.z; v[15] = (__bf16)d.w;
  return v;
}

// ---------------------------------------------------------------------------
// Kernel 1: Q = X (f32, [4096 x 768]) * Wq^T -> bf16. One 16x64 slab / wave.
// ---------------------------------------------------------------------------
__global__ void __launch_bounds__(128)
gemm_q_kernel(const float* __restrict__ X, const float* __restrict__ W,
              bf16_t* __restrict__ Cout) {
  const int lane = threadIdx.x & 31;
  const int wave = threadIdx.x >> 5;
  const int gw   = blockIdx.x * 4 + wave;     // 0..3071
  const int nSup = gw % (EMB / 64);
  const int mT   = gw / (EMB / 64);
  const int row0 = mT * 16, col0 = nSup * 64;
  const int half = lane >> 4, l16 = lane & 15;

  f32x8 acc[4] = {};
  const float* arow = X + (size_t)(row0 + l16) * EMB;
  for (int k = 0; k < EMB; k += 32) {
    bf16x16 af = frag_from_f32(arow + k, half);
#pragma unroll
    for (int nt = 0; nt < 4; ++nt) {
      const float* brow = W + (size_t)(col0 + nt * 16 + l16) * EMB + k;
      acc[nt] = wmma_bf16(af, frag_from_f32(brow, half), acc[nt]);
    }
  }
#pragma unroll
  for (int nt = 0; nt < 4; ++nt)
#pragma unroll
    for (int j = 0; j < 8; ++j) {
      const int r = row0 + j + 8 * half;
      const int c = col0 + nt * 16 + l16;
      Cout[(size_t)r * EMB + c] = (__bf16)acc[nt][j];
    }
}

// ---------------------------------------------------------------------------
// Kernel 2: KV = context * Wkv^T ([4096 x 1536]); cols [0,768) -> K bf16
// row-major; cols [768,1536) -> V transposed [b, 768, 2048] so attention V
// B-fragments are contiguous in the m (WMMA-K) dim.
// ---------------------------------------------------------------------------
__global__ void __launch_bounds__(128)
gemm_kv_kernel(const float* __restrict__ X, const float* __restrict__ W,
               bf16_t* __restrict__ Kout, bf16_t* __restrict__ Vt) {
  const int lane = threadIdx.x & 31;
  const int wave = threadIdx.x >> 5;
  const int gw   = blockIdx.x * 4 + wave;     // 0..6143
  const int nSup = gw % (2 * EMB / 64);
  const int mT   = gw / (2 * EMB / 64);
  const int row0 = mT * 16, col0 = nSup * 64;
  const int half = lane >> 4, l16 = lane & 15;

  f32x8 acc[4] = {};
  const float* arow = X + (size_t)(row0 + l16) * EMB;
  for (int k = 0; k < EMB; k += 32) {
    bf16x16 af = frag_from_f32(arow + k, half);
#pragma unroll
    for (int nt = 0; nt < 4; ++nt) {
      const float* brow = W + (size_t)(col0 + nt * 16 + l16) * EMB + k;
      acc[nt] = wmma_bf16(af, frag_from_f32(brow, half), acc[nt]);
    }
  }
  const bool isK = (col0 < EMB);   // uniform (col0 multiple of 64)
#pragma unroll
  for (int nt = 0; nt < 4; ++nt)
#pragma unroll
    for (int j = 0; j < 8; ++j) {
      const int r = row0 + j + 8 * half;
      const int c = col0 + nt * 16 + l16;
      const __bf16 v = (__bf16)acc[nt][j];
      if (isK) {
        Kout[(size_t)r * EMB + c] = v;
      } else {
        const int c2 = c - EMB, bb = r >> 11, mm = r & 2047;
        Vt[((size_t)bb * EMB + c2) * SEQ + mm] = v;
      }
    }
}

// ---------------------------------------------------------------------------
// Kernel 3: flash attention. Block = 4 waves sharing one (b,h); each wave
// owns one 16-row q-tile. Per 32-m chunk: K (32x64) and V^T (64x32) are
// DMA'd to LDS once per block (async, double-buffered), then each wave does
// S = QK^T (4 WMMA) -> scale+bias -> online softmax -> P relayout via LDS ->
// P*V (4 WMMA). Bias (402 MB, the bandwidth bound) is streamed exactly once.
// ---------------------------------------------------------------------------
__global__ void __launch_bounds__(128)
attn_kernel(const bf16_t* __restrict__ Qb, const bf16_t* __restrict__ Kb,
            const bf16_t* __restrict__ Vt, const float* __restrict__ bias,
            bf16_t* __restrict__ Ob) {
  __shared__ bf16_t kbuf[2][32][64];   // [buf][m][d]   4 KB each
  __shared__ bf16_t vbuf[2][64][32];   // [buf][d][m]   4 KB each
  __shared__ bf16_t ldsP[4][16 * 32];  // per-wave P relayout, 4 KB total

  const int tid  = threadIdx.x;
  const int lane = tid & 31;
  const int wave = tid >> 5;
  const int bh   = blockIdx.x >> 5;          // 24 (b,h) pairs, 32 blocks each
  const int qt   = (blockIdx.x & 31) * 4 + wave;
  const int h    = bh % HEADS, bb = bh / HEADS;
  const int half = lane >> 4, l16 = lane & 15;
  const int q0   = qt * 16;

  const bf16_t* kbase = Kb + (size_t)bb * SEQ * EMB + h * HDIM;
  const bf16_t* vbase = Vt + ((size_t)bb * EMB + h * HDIM) * SEQ;

  // Stage one 32-m chunk into LDS: 4 async b128 instructions per wave.
  auto stage = [&](int m0, int bufsel) {
#pragma unroll
    for (int i = 0; i < 2; ++i) {              // K: 256 16B segments
      const int s = tid * 2 + i;
      const int row = s >> 3, seg = s & 7;
      lds_copy16(&kbuf[bufsel][row][seg * 8],
                 kbase + (size_t)(m0 + row) * EMB + seg * 8);
    }
#pragma unroll
    for (int i = 0; i < 2; ++i) {              // V^T: 256 16B segments
      const int s = tid * 2 + i;
      const int row = s >> 2, seg = s & 3;
      lds_copy16(&vbuf[bufsel][row][seg * 8],
                 vbase + (size_t)row * SEQ + m0 + seg * 8);
    }
  };

  // Q fragments for the two K-steps covering d=64 (loaded once)
  const bf16_t* qrow = Qb + ((size_t)bb * SEQ + q0 + l16) * EMB + h * HDIM;
  const bf16x16 qf0 = frag_from_bf16(qrow, half);
  const bf16x16 qf1 = frag_from_bf16(qrow + 32, half);

  float rowmax[8], rowsum[8];
  f32x8 o[4] = {};
#pragma unroll
  for (int j = 0; j < 8; ++j) { rowmax[j] = -1e30f; rowsum[j] = 0.f; }

  const float  scale    = 0.125f;  // 1/sqrt(64)
  const float* biasBase = bias + ((size_t)bb * HEADS + h) * SEQ * SEQ;
  bf16_t* myP = &ldsP[wave][0];

  stage(0, 0);
  for (int it = 0; it < SEQ / 32; ++it) {
    const int m0 = it * 32, bsel = it & 1;
    if (it + 1 < SEQ / 32) {
      stage(m0 + 32, bsel ^ 1);
      wait_async<4>();            // own current-chunk DMA retired (per-wave cnt)
    } else {
      wait_async<0>();
    }
    __syncthreads();              // all waves' chunk data visible in LDS

    // ---- S = Q K^T for two 16-column m sub-tiles (K from LDS)
    f32x8 s[2];
#pragma unroll
    for (int mt = 0; mt < 2; ++mt) {
      const bf16_t* krow = &kbuf[bsel][mt * 16 + l16][0];
      f32x8 z = {};
      z = wmma_bf16(qf0, frag_from_bf16(krow, half), z);
      z = wmma_bf16(qf1, frag_from_bf16(krow + 32, half), z);
      s[mt] = z;
    }
    // ---- scale + bias, running row-max (this lane's rows: j + 8*half)
    float nm[8], corr[8];
#pragma unroll
    for (int j = 0; j < 8; ++j) {
      const int r = j + 8 * half;
      const float* bp = biasBase + (size_t)(q0 + r) * SEQ + m0;
      const float s0 = s[0][j] * scale + bp[l16];
      const float s1 = s[1][j] * scale + bp[16 + l16];
      s[0][j] = s0; s[1][j] = s1;
      float v = fmaxf(s0, s1);
      v = fmaxf(v, __shfl_xor(v, 1, 32));
      v = fmaxf(v, __shfl_xor(v, 2, 32));
      v = fmaxf(v, __shfl_xor(v, 4, 32));
      v = fmaxf(v, __shfl_xor(v, 8, 32));
      const float newm = fmaxf(rowmax[j], v);
      corr[j] = __expf(rowmax[j] - newm);
      rowmax[j] = newm;
      nm[j] = newm;
    }
    // ---- P = exp(S - m), row sums, spill P (C-layout) to LDS as bf16
#pragma unroll
    for (int j = 0; j < 8; ++j) {
      const int r = j + 8 * half;
      const float p0 = __expf(s[0][j] - nm[j]);
      const float p1 = __expf(s[1][j] - nm[j]);
      myP[r * 32 + l16]      = (__bf16)p0;
      myP[r * 32 + 16 + l16] = (__bf16)p1;
      float ps = p0 + p1;
      ps += __shfl_xor(ps, 1, 32);
      ps += __shfl_xor(ps, 2, 32);
      ps += __shfl_xor(ps, 4, 32);
      ps += __shfl_xor(ps, 8, 32);
      rowsum[j] = rowsum[j] * corr[j] + ps;
    }
    __builtin_amdgcn_wave_barrier();
    asm volatile("" ::: "memory");  // keep LDS store->load order (DS ops are
                                    // in-order per wave in hardware)
    // ---- reload P in A-fragment layout (16x32, WMMA-K = m-chunk)
    const bf16x16 pf = frag_from_bf16(myP + l16 * 32, half);
    // ---- rescale O, accumulate P*V (V^T rows from LDS, m-contiguous)
#pragma unroll
    for (int nt = 0; nt < 4; ++nt) {
#pragma unroll
      for (int j = 0; j < 8; ++j) o[nt][j] *= corr[j];
      const bf16_t* vrow = &vbuf[bsel][nt * 16 + l16][0];
      o[nt] = wmma_bf16(pf, frag_from_bf16(vrow, half), o[nt]);
    }
    __syncthreads();              // compute done before buffer is re-staged
  }
  // ---- normalize and emit O as bf16 [b*n, 768]
#pragma unroll
  for (int j = 0; j < 8; ++j) rowsum[j] = 1.0f / rowsum[j];
#pragma unroll
  for (int nt = 0; nt < 4; ++nt)
#pragma unroll
    for (int j = 0; j < 8; ++j) {
      const int r = q0 + j + 8 * half;
      const int c = h * HDIM + nt * 16 + l16;
      Ob[((size_t)bb * SEQ + r) * EMB + c] = (__bf16)(o[nt][j] * rowsum[j]);
    }
}

// ---------------------------------------------------------------------------
// Kernel 4: out = O (bf16) * Wout^T (f32) + b_out -> f32 d_out.
// ---------------------------------------------------------------------------
__global__ void __launch_bounds__(128)
gemm_out_kernel(const bf16_t* __restrict__ Ob, const float* __restrict__ W,
                const float* __restrict__ bvec, float* __restrict__ out) {
  const int lane = threadIdx.x & 31;
  const int wave = threadIdx.x >> 5;
  const int gw   = blockIdx.x * 4 + wave;    // 0..3071
  const int nSup = gw % (EMB / 64);
  const int mT   = gw / (EMB / 64);
  const int row0 = mT * 16, col0 = nSup * 64;
  const int half = lane >> 4, l16 = lane & 15;

  f32x8 acc[4] = {};
  const bf16_t* arow = Ob + (size_t)(row0 + l16) * EMB;
  for (int k = 0; k < EMB; k += 32) {
    bf16x16 af = frag_from_bf16(arow + k, half);
#pragma unroll
    for (int nt = 0; nt < 4; ++nt) {
      const float* brow = W + (size_t)(col0 + nt * 16 + l16) * EMB + k;
      acc[nt] = wmma_bf16(af, frag_from_f32(brow, half), acc[nt]);
    }
  }
#pragma unroll
  for (int nt = 0; nt < 4; ++nt)
#pragma unroll
    for (int j = 0; j < 8; ++j) {
      const int r = row0 + j + 8 * half;
      const int c = col0 + nt * 16 + l16;
      out[(size_t)r * EMB + c] = acc[nt][j] + bvec[c];
    }
}

// ---------------------------------------------------------------------------
extern "C" void kernel_launch(void* const* d_in, const int* in_sizes, int n_in,
                              void* d_out, int out_size, void* d_ws,
                              size_t ws_size, hipStream_t stream) {
  const float* x    = (const float*)d_in[0];  // [2,2048,768]
  const float* ctx  = (const float*)d_in[1];  // [2,2048,768]
  const float* bias = (const float*)d_in[2];  // [2,12,2048,2048]
  const float* Wq   = (const float*)d_in[3];  // [768,768]
  const float* Wkv  = (const float*)d_in[4];  // [1536,768]
  const float* Wout = (const float*)d_in[5];  // [768,768]
  const float* bout = (const float*)d_in[6];  // [768]
  float* out = (float*)d_out;

  const size_t slab = (size_t)BATCH * SEQ * EMB;  // 3,145,728 elems
  bf16_t* qb = (bf16_t*)d_ws;
  bf16_t* kb = qb + slab;
  bf16_t* vt = kb + slab;   // V transposed: [b, 768, 2048]
  bf16_t* ob = vt + slab;   // attention output, bf16
  (void)in_sizes; (void)n_in; (void)out_size; (void)ws_size;

  gemm_q_kernel  <<< 768, 128, 0, stream>>>(x,   Wq,  qb);
  gemm_kv_kernel <<<1536, 128, 0, stream>>>(ctx, Wkv, kb, vt);
  attn_kernel    <<< 768, 128, 0, stream>>>(qb, kb, vt, bias, ob);
  gemm_out_kernel<<< 768, 128, 0, stream>>>(ob, Wout, bout, out);
}
